// TreeMemory_35940286333187
// MI455X (gfx1250) — matrix operand order repaired and checked by reading, not compiled, not run on hardware
//
#include <hip/hip_runtime.h>
#include <hip/hip_bf16.h>

// ---------------------------------------------------------------------------
// TreeMemory on MI455X (gfx1250, wave32).
// Heavy GEMMs in bf16 WMMA (v_wmma_f32_16x16x32_bf16, f32 accumulate) fed by
// 128-bit LDS fragment loads; weight panels staged via the Tensor Data Mover
// in the aggregation hot loop; 1-query attentions algebraically factored.
// ---------------------------------------------------------------------------

#define DMODEL 256
#define HEADS 8
#define DH 32
#define TOK 32
#define FFDIM 1024
#define BQ 4096            // B*M = 8*512
#define SCALE 0.17677669529663687f   // 1/sqrt(32)
#define NEGV -1.0e9f

typedef __attribute__((ext_vector_type(16))) __bf16 v16bf;
typedef __attribute__((ext_vector_type(8)))  float  v8f;
typedef __attribute__((ext_vector_type(4)))  unsigned int u32x4;
typedef __attribute__((ext_vector_type(8)))  int i32x8;
typedef __attribute__((ext_vector_type(4)))  int i32x4;

__device__ inline v8f wmma_bf16(v16bf a, v16bf b, v8f c) {
  return __builtin_amdgcn_wmma_f32_16x16x32_bf16(false, a, false, b, (short)0, c,
                                                 false, false);
}

// A fragment: 16x32 tile (rows row0.., cols k0..) of row-major bf16 matrix.
// Per-lane data is two contiguous 8-element runs -> 2x 128-bit loads.
__device__ inline v16bf frag_a(const __bf16* X, int ld, int row0, int k0, int lane) {
  int r = row0 + (lane & 15);
  int half = lane >> 4;
  const __bf16* p = X + (size_t)r * ld + k0 + half * 8;
  union { v16bf v; uint4 q[2]; } u;
  u.q[0] = *(const uint4*)(p);
  u.q[1] = *(const uint4*)(p + 16);
  return u.v;
}

// B fragment from TRANSPOSED operand (XT is N x K row-major, ld = K stride).
// Lane<16: N=col0+lane, K=kbase+0..7,16..23; lane>=16: N=col0+lane-16, K=kbase+8..15,24..31.
// Per-lane data is two contiguous 8-element runs -> 2x 128-bit loads.
__device__ inline v16bf frag_bt(const __bf16* XT, int ldt, int col0, int kbase, int lane) {
  int n = col0 + (lane & 15);
  int kb = kbase + ((lane >> 4) << 3);
  const __bf16* p = XT + (size_t)n * ldt + kb;
  union { v16bf v; uint4 q[2]; } u;
  u.q[0] = *(const uint4*)(p);
  u.q[1] = *(const uint4*)(p + 16);
  return u.v;
}

// C/D 16x16 f32: VGPR v: lanes 0-15 -> (M=row0+v, N=lane); lanes 16-31 -> M=row0+8+v.
__device__ inline void store_c(float* Cp, int ld, int row0, int col0, int lane, v8f c) {
  int n = lane & 15;
  int mo = (lane >> 4) << 3;
#pragma unroll
  for (int v = 0; v < 8; ++v)
    Cp[(size_t)(row0 + mo + v) * ld + col0 + n] = c[v];
}

// Tensor Data Mover: DMA a 2D bf16 tile (tile_k contiguous elems x tile_n rows,
// row stride row_stride_elems) from global into LDS at lds_byte_addr.
// D# packing per CDNA5 ISA ch.8 (group0: count/lds/global/type; group1: dims).
__device__ inline void tdm_load_2d_bf16(unsigned lds_byte_addr, const __bf16* gptr,
                                        unsigned tile_k, unsigned tile_n,
                                        unsigned row_stride_elems) {
  unsigned long long ga = (unsigned long long)gptr;
  u32x4 g0;
  g0[0] = 1u;                                                  // count=1, user D#
  g0[1] = lds_byte_addr;                                       // lds_addr
  g0[2] = (unsigned)(ga & 0xffffffffu);                        // global_addr lo
  g0[3] = (unsigned)((ga >> 32) & 0x01ffffffu) | (2u << 30);   // ga hi | type=2
  i32x8 g1;
  g1[0] = (int)(1u << 16);                                     // data_size=1 (2B)
  g1[1] = (int)((tile_k & 0xffffu) << 16);                     // tensor_dim0 lo16
  g1[2] = (int)(((tile_k >> 16) & 0xffffu) |
                ((tile_n & 0xffffu) << 16));                   // tdim0 hi | tdim1 lo
  g1[3] = (int)(((tile_n >> 16) & 0xffffu) |
                ((tile_k & 0xffffu) << 16));                   // tdim1 hi | tile_dim0
  g1[4] = (int)(tile_n & 0xffffu);                             // tile_dim1 (tile_dim2=0)
  g1[5] = (int)row_stride_elems;                               // tensor_dim0_stride lo32
  g1[6] = 0;
  g1[7] = 0;
  i32x4 z4 = {0, 0, 0, 0};
  i32x8 z8 = {0, 0, 0, 0, 0, 0, 0, 0};
  __builtin_amdgcn_tensor_load_to_lds(g0, g1, z4, z4, z8, 0);
}

__device__ inline float block_sum256(float v, volatile float* sred, int tid) {
#pragma unroll
  for (int m = 16; m >= 1; m >>= 1) v += __shfl_xor(v, m, 32);
  if ((tid & 31) == 0) sred[tid >> 5] = v;
  __syncthreads();
  if (tid == 0) {
    float a = 0.f;
    for (int i = 0; i < 8; ++i) a += sred[i];
    sred[0] = a;
  }
  __syncthreads();
  float r = sred[0];
  __syncthreads();
  return r;
}

__device__ inline float gelu_tanh(float x) {
  const float c = 0.7978845608028654f;  // sqrt(2/pi)
  float t = tanhf(c * (x + 0.044715f * x * x * x));
  return 0.5f * x * (1.f + t);
}

// ---------------------------------------------------------------------------
// Elementwise helpers
// ---------------------------------------------------------------------------
// Transposing f32 -> bf16 weight conversion: in is R x C row-major,
// out is C x R row-major (out[c*R + r] = in[r*C + c]).
__global__ void __launch_bounds__(256) f32_to_bf16_T_kernel(const float* __restrict__ in,
                                                            __bf16* __restrict__ out,
                                                            int R, int C) {
  int i = blockIdx.x * 256 + threadIdx.x;
  if (i < R * C) {
    int r = i / C, c = i % C;
    out[(size_t)c * R + r] = (__bf16)in[i];
  }
}

__global__ void __launch_bounds__(256) gelu_bf16_kernel(const float* __restrict__ in,
                                                        __bf16* __restrict__ out, int n) {
  int i = blockIdx.x * 256 + threadIdx.x;
  if (i < n) out[i] = (__bf16)gelu_tanh(in[i]);
}

__global__ void __launch_bounds__(256) add_out_kernel(const float* __restrict__ x,
                                                      const float* __restrict__ f2,
                                                      float* __restrict__ out, int n) {
  int i = blockIdx.x * 256 + threadIdx.x;
  if (i < n) out[i] = x[i] + f2[i];
}

// LayerNorm over rows of 256; one row per 256-thread block; bf16 output.
__global__ void __launch_bounds__(256) ln_rows_kernel(const float* __restrict__ in,
                                                      __bf16* __restrict__ out) {
  __shared__ float sred[16];
  int r = blockIdx.x, tid = threadIdx.x;
  float v = in[(size_t)r * DMODEL + tid];
  float mean = block_sum256(v, sred, tid) * (1.f / DMODEL);
  float var = block_sum256(v * v, sred, tid) * (1.f / DMODEL) - mean * mean;
  out[(size_t)r * DMODEL + tid] = (__bf16)((v - mean) * rsqrtf(var + 1e-5f));
}

// x = pred + q; lnx = LN(x) as bf16. One row per block.
__global__ void __launch_bounds__(256) add_ln_kernel(const float* __restrict__ pred,
                                                     const float* __restrict__ qd,
                                                     float* __restrict__ x,
                                                     __bf16* __restrict__ lnx) {
  __shared__ float sred[16];
  int r = blockIdx.x, tid = threadIdx.x;
  size_t i = (size_t)r * DMODEL + tid;
  float v = pred[i] + qd[i];
  x[i] = v;
  float mean = block_sum256(v, sred, tid) * (1.f / DMODEL);
  float var = block_sum256(v * v, sred, tid) * (1.f / DMODEL) - mean * mean;
  lnx[i] = (__bf16)((v - mean) * rsqrtf(var + 1e-5f));
}

// ---------------------------------------------------------------------------
// Generic bf16 WMMA GEMM: C(f32 MxN) = A(bf16 MxK) @ B, with B supplied
// TRANSPOSED (BT is N x K). 256 threads (8 waves); 64x64 C tile; 32-wide
// K panels staged in LDS; prefetch hints for the next panels.
// M,N multiples of 64; K multiple of 32.
// ---------------------------------------------------------------------------
__global__ void __launch_bounds__(256) gemm_bf16_kernel(const __bf16* __restrict__ A,
                                                        const __bf16* __restrict__ BT,
                                                        float* __restrict__ C,
                                                        int M, int N, int K) {
  __shared__ __bf16 As[64 * 32];    // 64 rows x 32 k
  __shared__ __bf16 BsT[64 * 32];   // 64 cols(n) x 32 k
  int tid = threadIdx.x, lane = tid & 31, wave = tid >> 5;
  int m0 = blockIdx.y * 64, n0 = blockIdx.x * 64;
  v8f acc0 = {}, acc1 = {};

  for (int k0 = 0; k0 < K; k0 += 32) {
    __syncthreads();
    {  // stage A panel 64x32 and BT panel 64x32 (256 x uint4 each)
      int ra = tid >> 2, ca = (tid & 3) * 8;
      *(uint4*)&As[ra * 32 + ca] = *(const uint4*)&A[(size_t)(m0 + ra) * K + k0 + ca];
      *(uint4*)&BsT[ra * 32 + ca] = *(const uint4*)&BT[(size_t)(n0 + ra) * K + k0 + ca];
    }
    // prefetch next K panels into cache (global_prefetch_b8)
    int kn = k0 + 32;
    if (kn < K) {
      if (tid < 64) __builtin_prefetch(&A[(size_t)(m0 + tid) * K + kn], 0, 1);
      else if (tid < 128) __builtin_prefetch(&BT[(size_t)(n0 + tid - 64) * K + kn], 0, 1);
    }
    __syncthreads();
    {
      int st = wave;           // sub-tile 0
      int tm = st >> 2, tn = st & 3;
      v16bf a = frag_a(As, 32, tm * 16, 0, lane);
      v16bf b = frag_bt(BsT, 32, tn * 16, 0, lane);
      acc0 = wmma_bf16(a, b, acc0);
    }
    {
      int st = wave + 8;       // sub-tile 1
      int tm = st >> 2, tn = st & 3;
      v16bf a = frag_a(As, 32, tm * 16, 0, lane);
      v16bf b = frag_bt(BsT, 32, tn * 16, 0, lane);
      acc1 = wmma_bf16(a, b, acc1);
    }
  }
  {
    int st = wave, tm = st >> 2, tn = st & 3;
    store_c(C, N, m0 + tm * 16, n0 + tn * 16, lane, acc0);
  }
  {
    int st = wave + 8, tm = st >> 2, tn = st & 3;
    store_c(C, N, m0 + tm * 16, n0 + tn * 16, lane, acc1);
  }
}

// ---------------------------------------------------------------------------
// Aggregation: one 256-thread block per group of 32 tokens.
// agg = agg_token; 2 layers of { agg += MHA(ln(agg), ln(x)); agg += FF(ln(agg)) }.
// K/V projections (32x256 @ 256x256) via WMMA; weight K-panels DMA'd into LDS
// by the Tensor Data Mover; 1-token Q/O/FF paths via VALU.
// ---------------------------------------------------------------------------
__global__ void __launch_bounds__(256) aggregate_kernel(
    const float* __restrict__ x, float* __restrict__ out,
    const float* __restrict__ agg_token,
    const __bf16* __restrict__ WkT, const __bf16* __restrict__ WvT,
    const float* __restrict__ Wq, const float* __restrict__ Wo,
    const float* __restrict__ W1, const float* __restrict__ W2) {
  extern __shared__ char smem[];
  const int LDP = 260;  // padded f32 row stride (avoid bank conflicts)
  __bf16* s_lnx = (__bf16*)smem;               // 32x256 bf16           @ 0
  __bf16* s_bpT = (__bf16*)(smem + 16384);     // 256(n) x 32(k) panel  @ 16384
  float* s_kh   = (float*)(smem + 32768);      // 32xLDP
  float* s_vh   = s_kh + 32 * LDP;             // 32xLDP
  float* s_agg  = s_vh + 32 * LDP;             // 256
  float* s_lnag = s_agg + 256;                 // 256
  float* s_q    = s_lnag + 256;                // 256 (scaled qh)
  float* s_attn = s_q + 256;                   // 256
  float* s_o    = s_attn + 256;                // 256
  float* s_h1   = s_o + 256;                   // 1024
  float* s_red  = s_h1 + 1024;                 // 16
  const unsigned BP_LDS_ADDR = 16384u;         // dynamic LDS starts at 0 (no static LDS)

  int tid = threadIdx.x, lane = tid & 31, wave = tid >> 5;
  const float* xblk = x + (size_t)blockIdx.x * TOK * DMODEL;

  // LN of the 32 input tokens -> bf16 in LDS (layer-invariant).
  {
    int token = tid >> 3, sub = tid & 7;
    const float* xr = xblk + (size_t)token * DMODEL;
    float s = 0.f, ss = 0.f;
#pragma unroll
    for (int j = 0; j < 32; ++j) {
      float v = xr[sub * 32 + j];
      s += v; ss += v * v;
    }
#pragma unroll
    for (int m = 1; m < 8; m <<= 1) {
      s += __shfl_xor(s, m, 32);
      ss += __shfl_xor(ss, m, 32);
    }
    float mean = s * (1.f / DMODEL);
    float var = ss * (1.f / DMODEL) - mean * mean;
    float inv = rsqrtf(var + 1e-5f);
#pragma unroll
    for (int j = 0; j < 32; ++j)
      s_lnx[token * DMODEL + sub * 32 + j] = (__bf16)((xr[sub * 32 + j] - mean) * inv);
  }
  s_agg[tid] = agg_token[tid];
  __syncthreads();

  for (int l = 0; l < 2; ++l) {
    const size_t wo = (size_t)l * DMODEL * DMODEL;
    const size_t fo1 = (size_t)l * DMODEL * FFDIM;
    const size_t fo2 = (size_t)l * FFDIM * DMODEL;

    // ln(agg)
    float av = s_agg[tid];
    float mean = block_sum256(av, s_red, tid) * (1.f / DMODEL);
    float var = block_sum256(av * av, s_red, tid) * (1.f / DMODEL) - mean * mean;
    s_lnag[tid] = (av - mean) * rsqrtf(var + 1e-5f);
    __syncthreads();

    // qh = ln(agg) @ Wq  (scaled by 1/sqrt(dh))
    {
      float qv = 0.f;
      for (int d = 0; d < DMODEL; ++d) qv += s_lnag[d] * Wq[wo + (size_t)d * DMODEL + tid];
      s_q[tid] = qv * SCALE;
    }

    // kh / vh = ln(x) @ W via WMMA; transposed weight K-panels DMA'd by the TDM.
    int tm = (wave >= 4) ? 1 : 0;
    int tn0 = (wave & 3) * 4;
    for (int proj = 0; proj < 2; ++proj) {
      const __bf16* WT = (proj == 0 ? WkT : WvT) + wo;  // N x K layout, ld=256
      float* dst = (proj == 0 ? s_kh : s_vh);
      v8f acc[4] = {v8f{}, v8f{}, v8f{}, v8f{}};
      for (int k0 = 0; k0 < DMODEL; k0 += 32) {
        __syncthreads();  // previous panel fully consumed
        if (wave == 0) {
          // 2D tile: 32 contiguous k-elems x 256 n-rows, row stride 256.
          tdm_load_2d_bf16(BP_LDS_ADDR, WT + (size_t)k0, 32u, 256u, 256u);
          __builtin_amdgcn_s_wait_tensorcnt(0);
        }
        __syncthreads();  // panel visible to all waves
        v16bf a = frag_a(s_lnx, DMODEL, tm * 16, k0, lane);
#pragma unroll
        for (int i = 0; i < 4; ++i) {
          v16bf b = frag_bt(s_bpT, 32, (tn0 + i) * 16, 0, lane);
          acc[i] = wmma_bf16(a, b, acc[i]);
        }
      }
#pragma unroll
      for (int i = 0; i < 4; ++i)
        store_c(dst, LDP, tm * 16, (tn0 + i) * 16, lane, acc[i]);
      __syncthreads();
    }

    // scores + softmax (per head = per wave, 32 keys = 32 lanes)
    {
      int h = wave, k = lane;
      float s = 0.f;
#pragma unroll
      for (int e = 0; e < DH; ++e) s += s_q[h * DH + e] * s_kh[k * LDP + h * DH + e];
      float mx = s;
#pragma unroll
      for (int m = 16; m >= 1; m >>= 1) mx = fmaxf(mx, __shfl_xor(mx, m, 32));
      float p = __expf(s - mx);
      float sum = p;
#pragma unroll
      for (int m = 16; m >= 1; m >>= 1) sum += __shfl_xor(sum, m, 32);
      s_attn[tid] = p / sum;
    }
    __syncthreads();

    // o[d] = sum_k a[head(d)][k] * vh[k][d];  attn_out = o @ Wo;  agg += attn_out
    {
      int h = tid >> 5;
      float ov = 0.f;
#pragma unroll
      for (int k = 0; k < TOK; ++k) ov += s_attn[h * TOK + k] * s_vh[k * LDP + tid];
      s_o[tid] = ov;
    }
    __syncthreads();
    {
      float w = 0.f;
      for (int i = 0; i < DMODEL; ++i) w += s_o[i] * Wo[wo + (size_t)i * DMODEL + tid];
      s_agg[tid] += w;
    }
    __syncthreads();

    // FF: agg += gelu(ln(agg) @ W1) @ W2
    float av2 = s_agg[tid];
    float mean2 = block_sum256(av2, s_red, tid) * (1.f / DMODEL);
    float var2 = block_sum256(av2 * av2, s_red, tid) * (1.f / DMODEL) - mean2 * mean2;
    s_lnag[tid] = (av2 - mean2) * rsqrtf(var2 + 1e-5f);
    __syncthreads();
#pragma unroll
    for (int jj = 0; jj < 4; ++jj) {
      int f = tid + jj * 256;
      float h = 0.f;
      for (int d = 0; d < DMODEL; ++d) h += s_lnag[d] * W1[fo1 + (size_t)d * FFDIM + f];
      s_h1[f] = gelu_tanh(h);
    }
    __syncthreads();
    {
      float f2 = 0.f;
      for (int f = 0; f < FFDIM; ++f) f2 += s_h1[f] * W2[fo2 + (size_t)f * DMODEL + tid];
      s_agg[tid] += f2;
    }
    __syncthreads();
  }
  out[(size_t)blockIdx.x * DMODEL + tid] = s_agg[tid];
}

// ---------------------------------------------------------------------------
// Tree-descent selection: one 256-thread block per query.
// s[h,k] = sum_c keys[k,c] * u[c,h], u[c,h] = sum_e qh[h,e]*Wk[c,h*32+e]
// softmax per head -> mean over heads -> argmax -> next index.
// ---------------------------------------------------------------------------
__global__ void __launch_bounds__(256) select_kernel(
    const __bf16* __restrict__ lnkeys, const float* __restrict__ qh_all,
    const float* __restrict__ qWk, const int* __restrict__ prev_idx, int stage,
    int* __restrict__ sel_out, int* __restrict__ next_out) {
  extern __shared__ char smem[];
  float* s_qh = (float*)smem;          // 256
  float* s_u  = s_qh + 256;            // 8*256
  float* s_k  = s_u + 2048;            // 32*256
  float* s_a  = s_k + 32 * 256;        // 256
  int q = blockIdx.x, tid = threadIdx.x;
  int b = q >> 9;
  int prev = (stage == 0) ? 0 : prev_idx[q];
  int grp = (stage == 0) ? b : (b * 32 + prev);
  const __bf16* krows = lnkeys + (size_t)grp * TOK * DMODEL;

  s_qh[tid] = qh_all[(size_t)q * DMODEL + tid];
  for (int i = tid; i < TOK * DMODEL; i += 256) s_k[i] = (float)krows[i];
  __syncthreads();

  for (int h = 0; h < HEADS; ++h) {
    float u = 0.f;
#pragma unroll
    for (int e = 0; e < DH; ++e)
      u += s_qh[h * DH + e] * qWk[(size_t)tid * DMODEL + h * DH + e];
    s_u[h * 256 + tid] = u;
  }
  __syncthreads();

  int h = tid >> 5, k = tid & 31;
  float s = 0.f;
  for (int c = 0; c < DMODEL; ++c) s += s_k[k * DMODEL + c] * s_u[h * 256 + c];
  s *= SCALE;
  float mx = s;
#pragma unroll
  for (int m = 16; m >= 1; m >>= 1) mx = fmaxf(mx, __shfl_xor(mx, m, 32));
  float p = __expf(s - mx);
  float sum = p;
#pragma unroll
  for (int m = 16; m >= 1; m >>= 1) sum += __shfl_xor(sum, m, 32);
  s_a[tid] = p / sum;
  __syncthreads();

  if (tid < 32) {
    float w = 0.f;
#pragma unroll
    for (int hh = 0; hh < HEADS; ++hh) w += s_a[hh * 32 + tid];
    int idx = tid;
#pragma unroll
    for (int m = 16; m >= 1; m >>= 1) {
      float ow = __shfl_xor(w, m, 32);
      int oi = __shfl_xor(idx, m, 32);
      if (ow > w || (ow == w && oi < idx)) { w = ow; idx = oi; }
    }
    if (tid == 0) {
      sel_out[q] = idx;
      next_out[q] = (stage == 0) ? idx : (prev * 32 + idx);
    }
  }
}

// ---------------------------------------------------------------------------
// Final masked attention over 96 gathered keys; factored score/output math.
// Writes attention output o (pre-Wo) in f32 and bf16.
// ---------------------------------------------------------------------------
__global__ void __launch_bounds__(256) final_attn_kernel(
    const float* __restrict__ node, const __bf16* __restrict__ ln1,
    const __bf16* __restrict__ ln2, const int* __restrict__ sel0,
    const int* __restrict__ sel1, const int* __restrict__ next1,
    const int* __restrict__ next2, const float* __restrict__ qh_all,
    const float* __restrict__ qWk, const float* __restrict__ qWv,
    float* __restrict__ o_out, __bf16* __restrict__ o_bf) {
  extern __shared__ char smem[];
  const int LDK = 260;
  float* s_k  = (float*)smem;          // 96xLDK
  float* s_u  = s_k + 96 * LDK;        // 8*256
  float* s_a  = s_u + 2048;            // 8*96
  float* s_t  = s_a + 768;             // 8*256
  float* s_qh = s_t + 2048;            // 256
  int q = blockIdx.x, tid = threadIdx.x, b = q >> 9;

  s_qh[tid] = qh_all[(size_t)q * DMODEL + tid];
  // Keys 0..31: top level (per-batch); 32..63: selected mid group; 64..95: leaves (LN here).
  const __bf16* k2 = ln2 + (size_t)b * TOK * DMODEL;
  const __bf16* k1 = ln1 + (size_t)(b * 32 + next1[q]) * TOK * DMODEL;
  for (int i = tid; i < TOK * DMODEL; i += 256) {
    int r = i >> 8, c = i & 255;
    s_k[r * LDK + c] = (float)k2[i];
    s_k[(32 + r) * LDK + c] = (float)k1[i];
  }
  {
    int token = tid >> 3, sub = tid & 7;
    const float* xr = node + ((size_t)(b * 1024 + next2[q]) * TOK + token) * DMODEL;
    float s = 0.f, ss = 0.f;
#pragma unroll
    for (int j = 0; j < 32; ++j) {
      float v = xr[sub * 32 + j];
      s += v; ss += v * v;
    }
#pragma unroll
    for (int m = 1; m < 8; m <<= 1) {
      s += __shfl_xor(s, m, 32);
      ss += __shfl_xor(ss, m, 32);
    }
    float mean = s * (1.f / DMODEL);
    float var = ss * (1.f / DMODEL) - mean * mean;
    float inv = rsqrtf(var + 1e-5f);
#pragma unroll
    for (int j = 0; j < 32; ++j)
      s_k[(64 + token) * LDK + sub * 32 + j] = (xr[sub * 32 + j] - mean) * inv;
  }
  __syncthreads();

  for (int h = 0; h < HEADS; ++h) {
    float u = 0.f;
#pragma unroll
    for (int e = 0; e < DH; ++e)
      u += s_qh[h * DH + e] * qWk[(size_t)tid * DMODEL + h * DH + e];
    s_u[h * 256 + tid] = u;
  }
  __syncthreads();

  int h = tid >> 5, kl = tid & 31;
  int s0 = sel0[q], s1 = 32 + sel1[q];
  float sc[3];
  float mx = -1e30f;
#pragma unroll
  for (int j = 0; j < 3; ++j) {
    int k = kl + 32 * j;
    float s = 0.f;
    for (int c = 0; c < DMODEL; ++c) s += s_k[k * LDK + c] * s_u[h * 256 + c];
    s *= SCALE;
    if (k == s0 || k == s1) s = NEGV;
    sc[j] = s;
    mx = fmaxf(mx, s);
  }
#pragma unroll
  for (int m = 16; m >= 1; m >>= 1) mx = fmaxf(mx, __shfl_xor(mx, m, 32));
  float sum = 0.f;
#pragma unroll
  for (int j = 0; j < 3; ++j) { sc[j] = __expf(sc[j] - mx); sum += sc[j]; }
#pragma unroll
  for (int m = 16; m >= 1; m >>= 1) sum += __shfl_xor(sum, m, 32);
  float inv = 1.f / sum;
#pragma unroll
  for (int j = 0; j < 3; ++j) s_a[h * 96 + kl + 32 * j] = sc[j] * inv;
  __syncthreads();

  // t[h,c] = sum_k a[h,k]*keys[k,c];  o[d] = sum_c t[head(d),c]*Wv[c,d]
  for (int hh = 0; hh < HEADS; ++hh) {
    float t = 0.f;
    for (int k = 0; k < 96; ++k) t += s_a[hh * 96 + k] * s_k[k * LDK + tid];
    s_t[hh * 256 + tid] = t;
  }
  __syncthreads();
  int hd = tid >> 5;
  float o = 0.f;
  for (int c = 0; c < DMODEL; ++c) o += s_t[hd * 256 + c] * qWv[(size_t)c * DMODEL + tid];
  o_out[(size_t)q * DMODEL + tid] = o;
  o_bf[(size_t)q * DMODEL + tid] = (__bf16)o;
}

// ---------------------------------------------------------------------------
// Host orchestration
// ---------------------------------------------------------------------------
extern "C" void kernel_launch(void* const* d_in, const int* in_sizes, int n_in,
                              void* d_out, int out_size, void* d_ws, size_t ws_size,
                              hipStream_t stream) {
  (void)in_sizes; (void)n_in; (void)out_size; (void)ws_size;
  const float* query   = (const float*)d_in[0];
  const float* node    = (const float*)d_in[1];
  const float* aggtok  = (const float*)d_in[2];
  const float* aWq     = (const float*)d_in[3];
  const float* aWk     = (const float*)d_in[4];
  const float* aWv     = (const float*)d_in[5];
  const float* aWo     = (const float*)d_in[6];
  const float* aW1     = (const float*)d_in[7];
  const float* aW2     = (const float*)d_in[8];
  const float* qWq     = (const float*)d_in[9];
  const float* qWk     = (const float*)d_in[10];
  const float* qWv     = (const float*)d_in[11];
  const float* qWo     = (const float*)d_in[12];
  const float* fW1     = (const float*)d_in[13];
  const float* fW2     = (const float*)d_in[14];
  float* outp = (float*)d_out;

  char* w = (char*)d_ws;
  size_t off = 0;
  auto nxt = [&](size_t bytes) -> char* {
    char* p = w + off;
    off += (bytes + 255) & ~(size_t)255;
    return p;
  };
  __bf16* aWkT = (__bf16*)nxt(2 * 256 * 256 * 2);   // per-layer transposed (NxK)
  __bf16* aWvT = (__bf16*)nxt(2 * 256 * 256 * 2);
  __bf16* qWqT = (__bf16*)nxt(256 * 256 * 2);
  __bf16* qWoT = (__bf16*)nxt(256 * 256 * 2);
  __bf16* fW1T = (__bf16*)nxt(256 * 1024 * 2);      // 1024 x 256
  __bf16* fW2T = (__bf16*)nxt(1024 * 256 * 2);      // 256 x 1024
  float*  level1 = (float*)nxt((size_t)8192 * 256 * 4);
  float*  level2 = (float*)nxt((size_t)256 * 256 * 4);
  __bf16* ln1_b  = (__bf16*)nxt((size_t)8192 * 256 * 2);
  __bf16* ln2_b  = (__bf16*)nxt((size_t)256 * 256 * 2);
  __bf16* qn_b   = (__bf16*)nxt((size_t)BQ * 256 * 2);
  float*  qh     = (float*)nxt((size_t)BQ * 256 * 4);
  int* sel0  = (int*)nxt(BQ * 4);
  int* sel1  = (int*)nxt(BQ * 4);
  int* next1 = (int*)nxt(BQ * 4);
  int* next2 = (int*)nxt(BQ * 4);
  float*  o_f  = (float*)nxt((size_t)BQ * 256 * 4);
  __bf16* o_b  = (__bf16*)nxt((size_t)BQ * 256 * 2);
  float*  pred = (float*)nxt((size_t)BQ * 256 * 4);
  float*  xbuf = (float*)nxt((size_t)BQ * 256 * 4);
  __bf16* lnx_b = (__bf16*)nxt((size_t)BQ * 256 * 2);
  float*  h1   = (float*)nxt((size_t)BQ * 1024 * 4);
  __bf16* h1g_b = (__bf16*)nxt((size_t)BQ * 1024 * 2);
  float*  f2   = (float*)nxt((size_t)BQ * 256 * 4);

  const int AGG_SMEM = 32768 + 2 * 32 * 260 * 4 + (5 * 256 + 1024 + 16) * 4;
  const int SEL_SMEM = (256 + 2048 + 32 * 256 + 256) * 4;
  const int FIN_SMEM = (96 * 260 + 2048 + 768 + 2048 + 256) * 4;
  (void)hipFuncSetAttribute((const void*)aggregate_kernel,
                            hipFuncAttributeMaxDynamicSharedMemorySize, AGG_SMEM);
  (void)hipFuncSetAttribute((const void*)select_kernel,
                            hipFuncAttributeMaxDynamicSharedMemorySize, SEL_SMEM);
  (void)hipFuncSetAttribute((const void*)final_attn_kernel,
                            hipFuncAttributeMaxDynamicSharedMemorySize, FIN_SMEM);

  // 1) Weight conversions to transposed bf16 (per layer for the 2-layer stacks).
  const int g64k = (65536 + 255) / 256, g256k = (262144 + 255) / 256;
  f32_to_bf16_T_kernel<<<g64k, 256, 0, stream>>>(aWk, aWkT, 256, 256);
  f32_to_bf16_T_kernel<<<g64k, 256, 0, stream>>>(aWk + 65536, aWkT + 65536, 256, 256);
  f32_to_bf16_T_kernel<<<g64k, 256, 0, stream>>>(aWv, aWvT, 256, 256);
  f32_to_bf16_T_kernel<<<g64k, 256, 0, stream>>>(aWv + 65536, aWvT + 65536, 256, 256);
  f32_to_bf16_T_kernel<<<g64k, 256, 0, stream>>>(qWq, qWqT, 256, 256);
  f32_to_bf16_T_kernel<<<g64k, 256, 0, stream>>>(qWo, qWoT, 256, 256);
  f32_to_bf16_T_kernel<<<g256k, 256, 0, stream>>>(fW1, fW1T, 256, 1024);
  f32_to_bf16_T_kernel<<<g256k, 256, 0, stream>>>(fW2, fW2T, 1024, 256);

  // 2) qn = LN(query); qh = qn @ qWq  (WMMA GEMM 4096x256x256).
  ln_rows_kernel<<<BQ, 256, 0, stream>>>(query, qn_b);
  gemm_bf16_kernel<<<dim3(256 / 64, BQ / 64), 256, 0, stream>>>(qn_b, qWqT, qh,
                                                                BQ, 256, 256);

  // 3) Tree aggregation (leaves -> level1 -> level2), WMMA K/V projections + TDM.
  aggregate_kernel<<<8192, 256, AGG_SMEM, stream>>>(node, level1, aggtok, aWkT, aWvT,
                                                    aWq, aWo, aW1, aW2);
  aggregate_kernel<<<256, 256, AGG_SMEM, stream>>>(level1, level2, aggtok, aWkT, aWvT,
                                                   aWq, aWo, aW1, aW2);

  // 4) LN of tree levels for key gathering.
  ln_rows_kernel<<<8192, 256, 0, stream>>>(level1, ln1_b);
  ln_rows_kernel<<<256, 256, 0, stream>>>(level2, ln2_b);

  // 5) Descent selections.
  select_kernel<<<BQ, 256, SEL_SMEM, stream>>>(ln2_b, qh, qWk, nullptr, 0, sel0, next1);
  select_kernel<<<BQ, 256, SEL_SMEM, stream>>>(ln1_b, qh, qWk, next1, 1, sel1, next2);

  // 6) Final masked attention (factored), then pred = o @ qWo (WMMA).
  final_attn_kernel<<<BQ, 256, FIN_SMEM, stream>>>(node, ln1_b, ln2_b, sel0, sel1,
                                                   next1, next2, qh, qWk, qWv, o_f, o_b);
  gemm_bf16_kernel<<<dim3(256 / 64, BQ / 64), 256, 0, stream>>>(o_b, qWoT, pred,
                                                                BQ, 256, 256);

  // 7) x = pred + query; FF: out = x + gelu(LN(x) @ fW1) @ fW2 (WMMA GEMMs).
  add_ln_kernel<<<BQ, 256, 0, stream>>>(pred, query, xbuf, lnx_b);
  gemm_bf16_kernel<<<dim3(1024 / 64, BQ / 64), 256, 0, stream>>>(lnx_b, fW1T, h1,
                                                                 BQ, 1024, 256);
  gelu_bf16_kernel<<<((BQ * 1024) + 255) / 256, 256, 0, stream>>>(h1, h1g_b, BQ * 1024);
  gemm_bf16_kernel<<<dim3(256 / 64, BQ / 64), 256, 0, stream>>>(h1g_b, fW2T, f2,
                                                                BQ, 256, 1024);
  add_out_kernel<<<((BQ * 256) + 255) / 256, 256, 0, stream>>>(xbuf, f2, outp, BQ * 256);
}
